// Router_35356170781338
// MI455X (gfx1250) — matrix operand order repaired and checked by reading, not compiled, not run on hardware
//
#include <hip/hip_runtime.h>
#include <math.h>
#include <stdint.h>

// ---------------------------------------------------------------------------
// Router kernel suite for MI455X (gfx1250, wave32, WMMA, TDM).
//
//   1) prep_neurons : neuron_pos f32 -> f16 copy + |n|^2 norms
//   2) prep_x       : x f32 -> f16 copy (WMMA A operand source)
//   3) prep_wt      : W_p{qk,v,kn} (1024x64) -> f16 transposed (64x1024)
//   4) pos_proj     : posH[set] = f16(x @ W + b)     (WMMA f16 16x16x32)
//   5) tau_proj     : 4 scalar thresholds per token
//   6) dist_topk    : WMMA distance tiles (|n|^2 - 2 p.n) + per-chunk top-32
//   7) merge_topk   : merge chunk lists -> sorted top-32 -> ci outputs
//   8) score_gate   : gather-dot s = (x/0.9).neuron, wave32 threshold gate
//
// LDS staging of contiguous tiles goes through the Tensor Data Mover when the
// toolchain exposes __builtin_amdgcn_tensor_load_to_lds (s_wait_tensorcnt to
// sync), with a plain vectorized-copy fallback.
// ---------------------------------------------------------------------------

typedef __attribute__((ext_vector_type(16))) _Float16 v16h;
typedef __attribute__((ext_vector_type(8)))  _Float16 v8h;
typedef __attribute__((ext_vector_type(8)))  float    v8f;

union frag16 { v16h v; v8h h[2]; };
union h4pack { _Float16 h[4]; uint2 u; };

#define BS      4096      // B*S tokens
#define DMODEL  1024
#define PDIM    64
#define NQK     4096
#define NV      4096
#define NKNOW   8192
#define NTOT    16384
#define CHUNK   512
#define KSEL    32        // MAX_K*2 candidates kept
#define SENT    1e30f

// ---------------------------------------------------------------------------
// Tensor Data Mover 1-D stage helper (guarded; falls back to plain copies)
// ---------------------------------------------------------------------------
#if defined(__has_builtin)
#if __has_builtin(__builtin_amdgcn_tensor_load_to_lds) && \
    __has_builtin(__builtin_amdgcn_s_wait_tensorcnt)
#define HAVE_TDM 1
#endif
#endif
#ifndef HAVE_TDM
#define HAVE_TDM 0
#endif

#if HAVE_TDM
typedef unsigned int u32x4 __attribute__((ext_vector_type(4)));
typedef int          i32x4 __attribute__((ext_vector_type(4)));
typedef int          i32x8 __attribute__((ext_vector_type(8)));

// Copy nelem contiguous elements (elem size code: 1 = 2B, 2 = 4B) from global
// to LDS using the TDM. Issue from a single wave; EXEC is ignored by TDM.
__device__ __forceinline__ void tdm_stage_1d(const void* gsrc, void* lds,
                                             unsigned ds_code, unsigned nelem) {
  unsigned long long ga = (unsigned long long)(uintptr_t)gsrc;
  unsigned la = (unsigned)(uintptr_t)lds;     // LDS byte addr = addr[31:0]
  u32x4 g0;
  g0[0] = 1u;                                 // count=1, user descriptor
  g0[1] = la;                                 // lds_addr
  g0[2] = (unsigned)ga;                       // global_addr[31:0]
  g0[3] = (unsigned)(ga >> 32) | (2u << 30);  // global_addr[56:32] | type=2
  i32x8 g1 = {};
  g1[0] = (int)(ds_code << 16);               // data_size (bits 17:16)
  g1[1] = (int)(nelem << 16);                 // tensor_dim0[15:0] (bits 63:48)
  g1[2] = (int)(1u << 16);                    // tensor_dim1 = 1  (bits 95:80)
  g1[3] = (int)(nelem << 16);                 // tile_dim0 (bits 127:112)
  g1[4] = 1;                                  // tile_dim1 = 1 (bits 143:128)
  g1[5] = (int)nelem;                         // tensor_dim0_stride (low 32)
  i32x4 z4 = {};
#if __clang_major__ <= 22
  __builtin_amdgcn_tensor_load_to_lds(g0, g1, z4, z4, 0);
#else
  i32x8 z8 = {};
  __builtin_amdgcn_tensor_load_to_lds(g0, g1, z4, z4, z8, 0);
#endif
  __builtin_amdgcn_s_wait_tensorcnt(0);
}
#endif

__device__ __forceinline__ float wave_red_add(float v) {
#pragma unroll
  for (int o = 16; o; o >>= 1) v += __shfl_xor(v, o);
  return v;
}

// ---------------------------------------------------------------------------
// 1) neuron_pos -> f16 + squared norms
// ---------------------------------------------------------------------------
__global__ __launch_bounds__(256) void prep_neurons(
    const float* __restrict__ npos, _Float16* __restrict__ nposH,
    float* __restrict__ nnorm) {
  int n = blockIdx.x * blockDim.x + threadIdx.x;
  if (n >= NTOT) return;
  const float* r = npos + (size_t)n * PDIM;
  uint2* outp = (uint2*)(nposH + (size_t)n * PDIM);
  float s = 0.f;
#pragma unroll
  for (int i = 0; i < PDIM; i += 4) {
    float4 v = *(const float4*)(r + i);
    h4pack p;
    p.h[0] = (_Float16)v.x; p.h[1] = (_Float16)v.y;
    p.h[2] = (_Float16)v.z; p.h[3] = (_Float16)v.w;
    outp[i >> 2] = p.u;
    s += v.x * v.x + v.y * v.y + v.z * v.z + v.w * v.w;
  }
  nnorm[n] = s;
}

// ---------------------------------------------------------------------------
// 2) x -> f16 (WMMA A operand source), 4 elements per thread
// ---------------------------------------------------------------------------
__global__ __launch_bounds__(256) void prep_x(
    const float* __restrict__ x, _Float16* __restrict__ xH) {
  size_t i = (size_t)blockIdx.x * blockDim.x + threadIdx.x;  // over 1M float4s
  float4 v = *(const float4*)(x + i * 4);
  h4pack p;
  p.h[0] = (_Float16)v.x; p.h[1] = (_Float16)v.y;
  p.h[2] = (_Float16)v.z; p.h[3] = (_Float16)v.w;
  ((uint2*)xH)[i] = p.u;
}

// ---------------------------------------------------------------------------
// 3) W_p (1024x64) -> f16 transpose WT[set][64][1024]
// ---------------------------------------------------------------------------
__global__ __launch_bounds__(256) void prep_wt(
    const float* __restrict__ W0, const float* __restrict__ W1,
    const float* __restrict__ W2, _Float16* __restrict__ WT) {
  int i = blockIdx.x * blockDim.x + threadIdx.x;  // over 3*64*1024
  int set = i >> 16, rem = i & 65535;
  int n = rem >> 10, k = rem & 1023;
  const float* W = (set == 0) ? W0 : ((set == 1) ? W1 : W2);
  WT[i] = (_Float16)W[(size_t)k * PDIM + n];
}

// ---------------------------------------------------------------------------
// 4) position projections: posH[set] = f16(x @ W + b)
//    block = 128 (4 waves), grid = (256 token tiles, 3 sets);
//    wave w owns output columns [16w, 16w+16). A tile staged once (32 KB).
// ---------------------------------------------------------------------------
__global__ __launch_bounds__(128) void pos_proj(
    const _Float16* __restrict__ xH, const _Float16* __restrict__ WT,
    const float* __restrict__ b0, const float* __restrict__ b1,
    const float* __restrict__ b2, _Float16* __restrict__ posH) {
  const int set  = blockIdx.y;
  const float* bias = (set == 0) ? b0 : ((set == 1) ? b1 : b2);
  const int tile = blockIdx.x;
  const int tid  = threadIdx.x;
  const int lane = tid & 31;
  const int wave = tid >> 5;  // 0..3
  const int hi   = lane >> 4;
  const int mlo  = lane & 15;

  __shared__ _Float16 xs[16][DMODEL];  // 32 KB, row-major token tile

  const _Float16* xsrc = xH + (size_t)tile * 16 * DMODEL;
#if HAVE_TDM
  if (wave == 0) tdm_stage_1d(xsrc, &xs[0][0], 1, 16 * DMODEL);
#else
  for (int i = tid; i < 16 * DMODEL / 8; i += 128)
    ((uint4*)xs)[i] = ((const uint4*)xsrc)[i];
#endif
  __syncthreads();

  // B rows: WT[set][n][k], n = wave*16 + mlo; contiguous 32B per fragment.
  const _Float16* wrow = WT + ((size_t)set * PDIM + wave * 16 + mlo) * DMODEL;

  v8f acc = {};
  for (int kt = 0; kt < DMODEL / 32; ++kt) {
    // A fragment: lane covers K = kt*32 + hi*16 + {0..15}
    const _Float16* ar = &xs[mlo][kt * 32 + hi * 16];
    frag16 a, b;
    a.v = *(const v16h*)(ar);
    b.v = *(const v16h*)(wrow + kt * 32 + hi * 16);
    acc = __builtin_amdgcn_wmma_f32_16x16x32_f16(
        false, a.v, false, b.v, (short)0, acc, false, false);
  }
  const int col = wave * 16 + mlo;
  const float bc = bias[col];
  _Float16* out = posH + (size_t)set * BS * PDIM;
#pragma unroll
  for (int r = 0; r < 8; ++r) {
    int row = r + (hi ? 8 : 0);  // C/D layout
    out[(size_t)(tile * 16 + row) * PDIM + col] = (_Float16)(acc[r] + bc);
  }
}

// ---------------------------------------------------------------------------
// 5) tau: 4 scalar projections per token; one wave per token
// ---------------------------------------------------------------------------
__global__ __launch_bounds__(128) void tau_proj(
    const float* __restrict__ x,
    const float* __restrict__ Wta, const float* __restrict__ bta,
    const float* __restrict__ Wtk, const float* __restrict__ btk,
    float* __restrict__ tau) {
  const int wave = threadIdx.x >> 5, lane = threadIdx.x & 31;
  const int t = blockIdx.x * 4 + wave;
  const float* xr = x + (size_t)t * DMODEL;
  float a0 = 0.f, a1 = 0.f, a2 = 0.f, a3 = 0.f;
  for (int j = lane; j < DMODEL; j += 32) {
    float xv = xr[j];
    a0 += xv * Wta[j * 3 + 0];
    a1 += xv * Wta[j * 3 + 1];
    a2 += xv * Wta[j * 3 + 2];
    a3 += xv * Wtk[j];
  }
  a0 = wave_red_add(a0); a1 = wave_red_add(a1);
  a2 = wave_red_add(a2); a3 = wave_red_add(a3);
  if (lane == 0) {
    tau[t * 4 + 0] = a0 + bta[0];
    tau[t * 4 + 1] = a1 + bta[1];
    tau[t * 4 + 2] = a2 + bta[2];
    tau[t * 4 + 3] = a3 + btk[0];
  }
}

// ---------------------------------------------------------------------------
// 6) distance tiles + per-chunk top-32.
//    grid = (256 token tiles, 32 chunk-units: 8 qk + 8 v + 16 know)
//    block = 256 (8 waves). Ranking key: |n|^2 - 2 p.n.
// ---------------------------------------------------------------------------
__global__ __launch_bounds__(256) void dist_topk(
    const _Float16* __restrict__ posH, const _Float16* __restrict__ nposH,
    const float* __restrict__ nnorm, float2* __restrict__ ctop) {
  const int y = blockIdx.y;
  int set, chunk;
  if (y < 8)       { set = 0; chunk = y; }
  else if (y < 16) { set = 1; chunk = y - 8; }
  else             { set = 2; chunk = y - 16; }
  const int nch     = (set == 2) ? 16 : 8;
  const int setNOff = (set == 0) ? 0 : ((set == 1) ? NQK : NQK + NV);
  const size_t tbase =
      (set == 0) ? 0
                 : ((set == 1) ? (size_t)BS * 8 * KSEL
                               : (size_t)BS * 16 * KSEL);

  const int tile = blockIdx.x;
  const int tid  = threadIdx.x;
  const int lane = tid & 31;
  const int wave = tid >> 5;  // 0..7
  const int hi   = lane >> 4;
  const int mlo  = lane & 15;

  __shared__ _Float16 posA[16][PDIM];   // 2 KB, row-major
  __shared__ float    dist[16][CHUNK];  // 32 KB

  const _Float16* psrc = posH + ((size_t)set * BS + tile * 16) * PDIM;
#if HAVE_TDM
  if (wave == 0) tdm_stage_1d(psrc, &posA[0][0], 1, 16 * PDIM);
#else
  if (tid < 128) ((uint4*)posA)[tid] = ((const uint4*)psrc)[tid];
#endif
  __syncthreads();

  // A fragments for K = 0..31 and 32..63 (row-major 32B runs per lane).
  const _Float16* ar = &posA[mlo][hi * 16];
  frag16 a0, a1;
  a0.v = *(const v16h*)(ar);
  a1.v = *(const v16h*)(ar + 32);

  // 512 neurons per chunk; 8 waves x 16 neurons -> 4 iterations.
  for (int it = 0; it < 4; ++it) {
    const int ntile = it * 128 + wave * 16;
    const int nglob = setNOff + chunk * CHUNK + ntile;
    const _Float16* bp = nposH + (size_t)(nglob + mlo) * PDIM + hi * 16;
    frag16 b0, b1;
    b0.v = *(const v16h*)(bp);
    b1.v = *(const v16h*)(bp + 32);
    v8f c = {};
    c = __builtin_amdgcn_wmma_f32_16x16x32_f16(
        false, a0.v, false, b0.v, (short)0, c, false, false);
    c = __builtin_amdgcn_wmma_f32_16x16x32_f16(
        false, a1.v, false, b1.v, (short)0, c, false, false);
    const float nn = nnorm[nglob + mlo];
#pragma unroll
    for (int r = 0; r < 8; ++r) {
      int row = r + (hi ? 8 : 0);
      dist[row][ntile + mlo] = nn - 2.0f * c[r];
    }
  }
  __syncthreads();

  // Selection: wave w extracts top-32 for tokens 2w, 2w+1. One LDS read
  // pass; values cached in registers, winners removed via broadcast index.
  for (int tt = 0; tt < 2; ++tt) {
    const int m     = wave * 2 + tt;
    const int tglob = tile * 16 + m;
    float lv[CHUNK / 32];
#pragma unroll
    for (int j = 0; j < CHUNK / 32; ++j) lv[j] = dist[m][lane + 32 * j];
    float2* top = ctop + tbase + ((size_t)tglob * nch + chunk) * KSEL;
    for (int round = 0; round < KSEL; ++round) {
      float bv = SENT;
      int   bi = 0x7fffffff;
#pragma unroll
      for (int j = 0; j < CHUNK / 32; ++j) {  // ascending j => lowest idx tie
        if (lv[j] < bv) { bv = lv[j]; bi = lane + 32 * j; }
      }
#pragma unroll
      for (int o = 16; o; o >>= 1) {  // wave argmin, lowest index on ties
        float ov = __shfl_xor(bv, o);
        int   oi = __shfl_xor(bi, o);
        if (ov < bv || (ov == bv && oi < bi)) { bv = ov; bi = oi; }
      }
      if ((bi & 31) == lane) lv[bi >> 5] = SENT;  // remove winner
      if (lane == 0) {
        float2 e;
        e.x = bv;
        e.y = __int_as_float(chunk * CHUNK + bi);
        top[round] = e;
      }
    }
  }
}

// ---------------------------------------------------------------------------
// 7) merge chunk lists -> final sorted top-32 indices; one wave per
//    (token, set); lane owns one slot of each chunk list.
// ---------------------------------------------------------------------------
__global__ __launch_bounds__(32) void merge_topk(
    const float2* __restrict__ ctop,
    int* __restrict__ ci0, int* __restrict__ ci1, int* __restrict__ ci2) {
  const int t = blockIdx.x, set = blockIdx.y, lane = threadIdx.x;
  const int nch = (set == 2) ? 16 : 8;
  const size_t tbase =
      (set == 0) ? 0
                 : ((set == 1) ? (size_t)BS * 8 * KSEL
                               : (size_t)BS * 16 * KSEL);
  const float2* src = ctop + tbase + (size_t)t * nch * KSEL;
  int* out = ((set == 0) ? ci0 : ((set == 1) ? ci1 : ci2)) + (size_t)t * KSEL;

  float val[16];
  int   idx[16];
  for (int k = 0; k < nch; ++k) {
    float2 e = src[(size_t)k * KSEL + lane];
    val[k] = e.x;
    idx[k] = __float_as_int(e.y);
  }
  for (int round = 0; round < KSEL; ++round) {
    float bv = SENT;
    int   bi = 0x7fffffff, bk = -1;
    for (int k = 0; k < nch; ++k)
      if (val[k] < bv || (val[k] == bv && idx[k] < bi)) {
        bv = val[k]; bi = idx[k]; bk = k;
      }
    float rv = bv; int ri = bi;
#pragma unroll
    for (int o = 16; o; o >>= 1) {
      float ov = __shfl_xor(rv, o);
      int   oi = __shfl_xor(ri, o);
      if (ov < rv || (ov == rv && oi < ri)) { rv = ov; ri = oi; }
    }
    if (bk >= 0 && ri == bi) val[bk] = SENT;  // neuron idx unique per set
    if (lane == 0) out[round] = ri;
  }
}

// ---------------------------------------------------------------------------
// 8) gather-dot scores + threshold gates. One block (128 thr) per token.
//    Keep rule: eg_i kept iff #{j: eg_j > eg_i} < 16  <=>  eg_i >= 16th
//    largest (matches eg >= top_k(eg,16)[-1] incl. ties).
// ---------------------------------------------------------------------------
__global__ __launch_bounds__(128) void score_gate(
    const float* __restrict__ x,
    const float* __restrict__ qkN, const float* __restrict__ vN,
    const float* __restrict__ knN,
    const int* __restrict__ ci0, const int* __restrict__ ci1,
    const int* __restrict__ ci2,
    const float* __restrict__ tau,
    float* __restrict__ gQ, float* __restrict__ gK,
    float* __restrict__ gV, float* __restrict__ gKN) {
  const int t = blockIdx.x;
  const int tid = threadIdx.x, lane = tid & 31, wave = tid >> 5;  // 4 waves

  __shared__ float xs[DMODEL];
  __shared__ float sc[3][KSEL];

  const float* xsrc = x + (size_t)t * DMODEL;
#if HAVE_TDM
  if (wave == 0) tdm_stage_1d(xsrc, xs, 2, DMODEL);
#else
  for (int i = tid; i < DMODEL / 4; i += 128)
    *(float4*)(xs + i * 4) = *(const float4*)(xsrc + i * 4);
#endif
  __syncthreads();

  // 96 dot products of length 1024 (neuron tables are L2-resident: 64 MB)
  for (int d = wave; d < 96; d += 4) {
    const int set = d >> 5, cand = d & 31;
    const int* ci = (set == 0) ? ci0 : ((set == 1) ? ci1 : ci2);
    const float* NT = (set == 0) ? qkN : ((set == 1) ? vN : knN);
    const int nidx = ci[(size_t)t * KSEL + cand];
    const float* nr = NT + (size_t)nidx * DMODEL;
    __builtin_prefetch(nr, 0, 0);
    float acc = 0.f;
#pragma unroll
    for (int j = 0; j < 8; ++j) {
      int o = (j * 32 + lane) * 4;
      float4 a = *(const float4*)(xs + o);
      float4 b = *(const float4*)(nr + o);
      acc += a.x * b.x + a.y * b.y + a.z * b.z + a.w * b.w;
    }
    acc = wave_red_add(acc);
    if (lane == 0) sc[set][cand] = acc * (1.0f / 0.9f);  // x_drop scale
  }
  __syncthreads();

  // wave 0: g_Q(sQK,tau0) 1: g_K(sQK,tau1) 2: g_V(sV,tau2) 3: g_KN(sKN,tau3)
  const int srcset = (wave < 2) ? 0 : ((wave == 2) ? 1 : 2);
  const float s   = sc[srcset][lane];
  const float tv  = tau[(size_t)t * 4 + wave];
  const float raw = s - tv;
  const float gate = (raw > 0.f) ? raw : 1e-8f * expf(raw);
  const float eg   = expf(gate) - 1.0f;

  int cnt = 0;
#pragma unroll
  for (int j = 0; j < 32; ++j) {
    float o = __shfl(eg, j);
    cnt += (o > eg) ? 1 : 0;
  }
  const float kept = (cnt < 16) ? eg : 0.f;
  float sum = kept, mx = kept;
#pragma unroll
  for (int o = 16; o; o >>= 1) {
    sum += __shfl_xor(sum, o);
    float om = __shfl_xor(mx, o);
    mx = (om > mx) ? om : mx;
  }
  const float g = kept / (sum + 1e-8f) * tanhf(mx);
  float* outp = (wave == 0) ? gQ : ((wave == 1) ? gK : ((wave == 2) ? gV : gKN));
  outp[(size_t)t * KSEL + lane] = g;
}

// ---------------------------------------------------------------------------
extern "C" void kernel_launch(void* const* d_in, const int* in_sizes, int n_in,
                              void* d_out, int out_size, void* d_ws,
                              size_t ws_size, hipStream_t stream) {
  (void)in_sizes; (void)n_in; (void)out_size; (void)ws_size;

  const float* x     = (const float*)d_in[0];
  const float* npos  = (const float*)d_in[1];
  const float* W_pqk = (const float*)d_in[2];
  const float* b_pqk = (const float*)d_in[3];
  const float* W_pv  = (const float*)d_in[4];
  const float* b_pv  = (const float*)d_in[5];
  const float* W_pkn = (const float*)d_in[6];
  const float* b_pkn = (const float*)d_in[7];
  const float* W_ta  = (const float*)d_in[8];
  const float* b_ta  = (const float*)d_in[9];
  const float* W_tk  = (const float*)d_in[10];
  const float* b_tk  = (const float*)d_in[11];
  const float* qkN   = (const float*)d_in[12];
  const float* vN    = (const float*)d_in[13];
  const float* knN   = (const float*)d_in[14];

  // workspace layout (~44 MB total)
  char* w = (char*)d_ws;
  _Float16* xH    = (_Float16*)w;  w += (size_t)BS * DMODEL * 2;     // 8 MB
  _Float16* WT    = (_Float16*)w;  w += (size_t)3 * PDIM * DMODEL * 2;
  _Float16* posH  = (_Float16*)w;  w += (size_t)3 * BS * PDIM * 2;   // 1.5 MB
  _Float16* nposH = (_Float16*)w;  w += (size_t)NTOT * PDIM * 2;     // 2 MB
  float*    nnorm = (float*)w;     w += (size_t)NTOT * 4;
  float*    tau   = (float*)w;     w += (size_t)BS * 4 * 4;
  float2*   ctop  = (float2*)w;    // 4096*(8+8+16)*32 entries * 8B = 32 MB

  // outputs: g_Q g_K g_V g_KN (f32), ci_qk ci_v ci_kn (i32), each 4096*32
  float* out = (float*)d_out;
  float* gQ  = out;
  float* gK  = out + 131072;
  float* gV  = out + 262144;
  float* gKN = out + 393216;
  int* ci0 = (int*)(out + 524288);
  int* ci1 = (int*)(out + 655360);
  int* ci2 = (int*)(out + 786432);

  prep_neurons<<<NTOT / 256, 256, 0, stream>>>(npos, nposH, nnorm);
  prep_x<<<(BS * DMODEL / 4) / 256, 256, 0, stream>>>(x, xH);
  prep_wt<<<(3 * PDIM * DMODEL) / 256, 256, 0, stream>>>(
      W_pqk, W_pv, W_pkn, WT);
  pos_proj<<<dim3(BS / 16, 3), 128, 0, stream>>>(
      xH, WT, b_pqk, b_pv, b_pkn, posH);
  tau_proj<<<BS / 4, 128, 0, stream>>>(x, W_ta, b_ta, W_tk, b_tk, tau);
  dist_topk<<<dim3(BS / 16, 32), 256, 0, stream>>>(posH, nposH, nnorm, ctop);
  merge_topk<<<dim3(BS, 3), 32, 0, stream>>>(ctop, ci0, ci1, ci2);
  score_gate<<<BS, 128, 0, stream>>>(
      x, qkN, vN, knN, ci0, ci1, ci2, tau, gQ, gK, gV, gKN);
}
